// STAttentionBlock_89893665505905
// MI455X (gfx1250) — matrix operand order repaired and checked by reading, not compiled, not run on hardware
//
#include <hip/hip_runtime.h>
#include <math.h>

// ---------------- CDNA5 (gfx1250) types ----------------
typedef __bf16 bf16_t;
typedef __attribute__((ext_vector_type(16))) __bf16 v16bf;
typedef __attribute__((ext_vector_type(8)))  float  v8f;

union FragB16 { v16bf v; bf16_t e[16]; };
union FragF32 { v8f   v; float  e[8]; };

#define NN   128
#define CC   256
#define TT   32
#define VV   25
#define SS   3
#define ICC  64
#define PIX_T (TT * VV)           // 800 pixels per sample
#define P_TOT (NN * PIX_T)        // 102400 total pixels

__device__ __forceinline__ v8f wmma_bf16(v16bf a, v16bf b, v8f c) {
  // D = A(16x32 bf16) * B(32x16 bf16) + C(16x16 f32)
  return __builtin_amdgcn_wmma_f32_16x16x32_bf16(
      false, a, false, b, (short)0, c, false, false);
}

__device__ __forceinline__ void zero8(FragF32& f) {
#pragma unroll
  for (int i = 0; i < 8; ++i) f.e[i] = 0.f;
}

// ---- Fragment-ready ("swizzled") B layout -------------------------------
// Element (k, col) of a K x (16*ncoltiles) matrix lives at:
//   (((k>>5)*ncoltiles + (col>>4))*32 + ((col&15) | (((k>>4)&1)<<4)))*16 + (k&15)
// so a lane's 16 fragment elements are 32 contiguous bytes in LDS.
__device__ __forceinline__ int bswz(int k, int col, int ncoltiles) {
  return ((((k >> 5) * ncoltiles + (col >> 4)) * 32) +
          ((col & 15) | (((k >> 4) & 1) << 4))) * 16 + (k & 15);
}

// Vectorized B fragment read from swizzled LDS: two ds_load_b128.
__device__ __forceinline__ v16bf b_frag_swz(const bf16_t* base, int kstep,
                                            int coltile, int ncoltiles) {
  int lane = threadIdx.x & 31;
  const bf16_t* p = base + (((kstep * ncoltiles + coltile) * 32) + lane) * 16;
  FragB16 b;
  *(uint4*)&b.e[0] = ((const uint4*)p)[0];
  *(uint4*)&b.e[8] = ((const uint4*)p)[1];
  return b.v;
}

// A fragment (16x32) from a row-major [row][k] bf16 matrix (global), ld % 8 == 0,
// kbase % 32 == 0 -> two 16B-aligned global_load_b128 per lane.
__device__ __forceinline__ v16bf a_frag_rowmajor(const bf16_t* __restrict__ base,
                                                 int row, int ld, int kbase) {
  int lane = threadIdx.x & 31;
  const bf16_t* p = base + (size_t)(row + (lane & 15)) * ld + kbase + ((lane >> 4) << 4);
  FragB16 a;
  *(uint4*)&a.e[0] = *(const uint4*)p;         // K kh .. kh+7
  *(uint4*)&a.e[8] = *(const uint4*)(p + 8);   // K kh+8 .. kh+15
  return a.v;
}

// A fragment from row-major LDS with ld=32 (zero-padded): two ds_load_b128.
__device__ __forceinline__ v16bf a_frag_lds32(const bf16_t* base, int row, int kbase) {
  int lane = threadIdx.x & 31;
  const bf16_t* p = base + (row + (lane & 15)) * 32 + kbase + ((lane >> 4) << 4);
  FragB16 a;
  *(uint4*)&a.e[0] = *(const uint4*)p;
  *(uint4*)&a.e[8] = *(const uint4*)(p + 8);
  return a.v;
}

// ---------------- K1: transpose + positional encoding ----------------
__global__ void prep_acts(const float* __restrict__ x,
                          bf16_t* __restrict__ Xbf, bf16_t* __restrict__ Abf) {
  const float coef = -9.210340371976184f / (float)CC; // -ln(10000)/C
  size_t total = (size_t)NN * CC * TT * VV;
  for (size_t i = (size_t)blockIdx.x * blockDim.x + threadIdx.x; i < total;
       i += (size_t)gridDim.x * blockDim.x) {
    int v = (int)(i % VV); size_t r = i / VV;
    int t = (int)(r % TT); r /= TT;
    int c = (int)(r % CC); int n = (int)(r / CC);
    float xv = x[i];
    float div = __expf((float)(c & ~1) * coef);
    float ang = (float)v * div;
    float pe  = (c & 1) ? __cosf(ang) : __sinf(ang);
    size_t p = (size_t)n * PIX_T + t * VV + v;
    Xbf[(size_t)c * P_TOT + p] = (bf16_t)xv;
    Abf[(size_t)c * P_TOT + p] = (bf16_t)(xv + pe);
  }
}

// ---------------- K2: weight conversion + BN folding ----------------
__global__ void prep_weights(
    const float* __restrict__ in_w, const float* __restrict__ out_w,
    const float* __restrict__ d1_w, const float* __restrict__ d2_w,
    const float* __restrict__ ff_w,
    const float* __restrict__ out_b, const float* __restrict__ d1_b,
    const float* __restrict__ d2_b, const float* __restrict__ ff_b,
    const float* __restrict__ out_g, const float* __restrict__ out_be,
    const float* __restrict__ out_m, const float* __restrict__ out_v,
    const float* __restrict__ d1_g, const float* __restrict__ d1_be,
    const float* __restrict__ d1_m, const float* __restrict__ d1_v,
    const float* __restrict__ d2_g, const float* __restrict__ d2_be,
    const float* __restrict__ d2_m, const float* __restrict__ d2_v,
    const float* __restrict__ ff_g, const float* __restrict__ ff_be,
    const float* __restrict__ ff_m, const float* __restrict__ ff_v,
    bf16_t* __restrict__ inw, bf16_t* __restrict__ outw,
    bf16_t* __restrict__ d1w, bf16_t* __restrict__ d2w, bf16_t* __restrict__ ffw,
    float* __restrict__ scl, float* __restrict__ bia) {
  const size_t n_in = 384 * 256, n_out = 256 * 768, n_sq = 256 * 256;
  size_t total = n_in + n_out + 3 * n_sq;
  size_t gid = (size_t)blockIdx.x * blockDim.x + threadIdx.x;
  for (size_t i0 = gid; i0 < total; i0 += (size_t)gridDim.x * blockDim.x) {
    size_t i = i0;
    if (i < n_in)  { inw[i]  = (bf16_t)in_w[i];  continue; }
    i -= n_in;
    if (i < n_out) { outw[i] = (bf16_t)out_w[i]; continue; }
    i -= n_out;
    if (i < n_sq)  { d1w[i]  = (bf16_t)d1_w[i];  continue; }
    i -= n_sq;
    if (i < n_sq)  { d2w[i]  = (bf16_t)d2_w[i];  continue; }
    i -= n_sq;
    ffw[i] = (bf16_t)ff_w[i];
  }
  if (gid < 1024) {  // BN fold, order [out, d1, d2, ff]; conv bias folded in
    int which = (int)(gid >> 8), c = (int)(gid & 255);
    const float *g, *be, *m, *vv, *b;
    if (which == 0)      { g = out_g; be = out_be; m = out_m; vv = out_v; b = out_b; }
    else if (which == 1) { g = d1_g;  be = d1_be;  m = d1_m;  vv = d1_v;  b = d1_b; }
    else if (which == 2) { g = d2_g;  be = d2_be;  m = d2_m;  vv = d2_v;  b = d2_b; }
    else                 { g = ff_g;  be = ff_be;  m = ff_m;  vv = ff_v;  b = ff_b; }
    float s = g[c] * rsqrtf(vv[c] + 1e-5f);
    scl[which * 256 + c] = s;
    bia[which * 256 + c] = b[c] * s + be[c] - m[c] * s;
  }
}

// ---------------- K3: qk = in_w @ (x+pe) + in_b  (M=384,K=256,N=102400) ----------------
// Block tile: 128M x 64N. Each A fragment (2 global_load_b128) feeds 4 WMMAs.
__global__ __launch_bounds__(256) void qk_gemm(const bf16_t* __restrict__ Abf,
                                               const bf16_t* __restrict__ inw,
                                               const float* __restrict__ in_b,
                                               bf16_t* __restrict__ QK) {
  __shared__ __align__(16) bf16_t Bs[8 * 4 * 32 * 16];  // swizzled 256x64 panel (32KB)
  int bid = blockIdx.x;
  int mbase = (bid % 3) * 128;
  size_t pbase = (size_t)(bid / 3) * 64;
  { // stage + transpose-swizzle: one channel row (64 pixels, 128B) per thread
    int c = threadIdx.x;
    const uint4* src = (const uint4*)(Abf + (size_t)c * P_TOT + pbase);
    FragB16 row[4];
#pragma unroll
    for (int q = 0; q < 4; ++q) {
      *(uint4*)&row[q].e[0] = src[2 * q];
      *(uint4*)&row[q].e[8] = src[2 * q + 1];
    }
#pragma unroll
    for (int q = 0; q < 4; ++q)
#pragma unroll
      for (int j = 0; j < 16; ++j)
        Bs[bswz(c, q * 16 + j, 4)] = row[q].e[j];
  }
  __syncthreads();
  int wave = threadIdx.x >> 5;
  int row0 = mbase + wave * 16;
  FragF32 acc[4];
#pragma unroll
  for (int q = 0; q < 4; ++q) zero8(acc[q]);
#pragma unroll 1
  for (int kk = 0; kk < 8; ++kk) {
    __builtin_prefetch(inw + (size_t)row0 * 256 + kk * 32 + 32, 0, 0);
    v16bf a = a_frag_rowmajor(inw, row0, 256, kk * 32);
#pragma unroll
    for (int q = 0; q < 4; ++q) {
      v16bf b = b_frag_swz(Bs, kk, q, 4);
      acc[q].v = wmma_bf16(a, b, acc[q].v);
    }
  }
  int lane = threadIdx.x & 31;
#pragma unroll
  for (int q = 0; q < 4; ++q) {
    size_t p = pbase + q * 16 + (lane & 15);
#pragma unroll
    for (int r = 0; r < 8; ++r) {
      int o = row0 + r + ((lane >> 4) << 3);
      QK[(size_t)o * P_TOT + p] = (bf16_t)(acc[q].e[r] + in_b[o]);
    }
  }
}

// ---------------- K4: att = tanh(Q^T K / 2048)*alpha + att0 (one wave per (n,s)) ----------------
__global__ __launch_bounds__(256) void att_gemm(const bf16_t* __restrict__ QK,
                                                const float* __restrict__ alphas,
                                                const float* __restrict__ att0,
                                                float* __restrict__ attp) {
  int ws = blockIdx.x * 8 + (threadIdx.x >> 5);
  int n = ws / SS, s = ws % SS;
  int lane = threadIdx.x & 31;
  const size_t pbn = (size_t)n * PIX_T;
  const bf16_t* Qb = QK + (size_t)(s * ICC) * P_TOT + pbn;
  const bf16_t* Kb = QK + (size_t)((SS + s) * ICC) * P_TOT + pbn;
  FragF32 acc[2][2];
#pragma unroll
  for (int mt = 0; mt < 2; ++mt)
#pragma unroll
    for (int nt = 0; nt < 2; ++nt) zero8(acc[mt][nt]);

#pragma unroll 1
  for (int kk = 0; kk < 64; ++kk) {           // K = IC*T = 2048
    int kbase = kk * 32;
    FragB16 a[2], b[2];
#pragma unroll
    for (int mt = 0; mt < 2; ++mt) {          // A = Q^T : row u, K=(c,t)
      int u = mt * 16 + (lane & 15);
      bool ok = u < VV;
      int kh = kbase + ((lane >> 4) << 4);
#pragma unroll
      for (int j = 0; j < 8; ++j) {
        int k1 = kh + j;
        int k2 = kh + 8 + j;
        a[mt].e[j]     = ok ? Qb[(size_t)(k1 >> 5) * P_TOT + (k1 & 31) * VV + u] : (bf16_t)0.f;
        a[mt].e[8 + j] = ok ? Qb[(size_t)(k2 >> 5) * P_TOT + (k2 & 31) * VV + u] : (bf16_t)0.f;
      }
    }
#pragma unroll
    for (int nt = 0; nt < 2; ++nt) {          // B = K : row (c,t), col v
      int v = nt * 16 + (lane & 15);
      bool ok = v < VV;
      int k0 = kbase + ((lane >> 4) << 4);
#pragma unroll
      for (int j = 0; j < 16; ++j) {
        int k = k0 + j;
        b[nt].e[j] = ok ? Kb[(size_t)(k >> 5) * P_TOT + (k & 31) * VV + v] : (bf16_t)0.f;
      }
    }
#pragma unroll
    for (int mt = 0; mt < 2; ++mt)
#pragma unroll
      for (int nt = 0; nt < 2; ++nt)
        acc[mt][nt].v = wmma_bf16(a[mt].v, b[nt].v, acc[mt][nt].v);
  }
  float alpha = alphas[s];
  float* dst = attp + (size_t)(n * SS + s) * 32 * 32;  // zero-padded 32x32
#pragma unroll
  for (int mt = 0; mt < 2; ++mt)
#pragma unroll
    for (int nt = 0; nt < 2; ++nt)
#pragma unroll
      for (int r = 0; r < 8; ++r) {
        int u = mt * 16 + r + ((lane >> 4) << 3);
        int v = nt * 16 + (lane & 15);
        float val = 0.f;
        if (u < VV && v < VV)
          val = tanhf(acc[mt][nt].e[r] * (1.f / 2048.f)) * alpha +
                att0[(s * VV + u) * VV + v];
        dst[u * 32 + v] = val;
      }
}

// ---------------- K5: fused att-apply + out/d1/ff/d2 convs + BN + lrelu ----------------
// One block per (n,t); all intermediates stay in LDS (fragment-ready layouts).
__global__ __launch_bounds__(256) void fused_out(
    const bf16_t* __restrict__ Xbf, const float* __restrict__ attp,
    const bf16_t* __restrict__ outw, const bf16_t* __restrict__ d1w,
    const bf16_t* __restrict__ d2w, const bf16_t* __restrict__ ffw,
    const float* __restrict__ scl, const float* __restrict__ bia,
    float* __restrict__ out) {
  __shared__ __align__(16) bf16_t xsA[256 * 32];           // x, row-major ld32 (A operand)
  __shared__ __align__(16) bf16_t xsB[8 * 2 * 32 * 16];    // x, swizzled (B operand)
  __shared__ __align__(16) bf16_t attsB[SS * 2 * 32 * 16]; // att_s, swizzled
  __shared__ __align__(16) bf16_t y2sB[24 * 2 * 32 * 16];  // y2, swizzled; reused as u
  int n = blockIdx.x >> 5, t = blockIdx.x & 31;
  int tid = threadIdx.x, lane = tid & 31, wave = tid >> 5;
  size_t pbase = (size_t)n * PIX_T + t * VV;

  { // load x slice into both layouts (one channel row per thread)
    int c = tid;
    const bf16_t* src = Xbf + (size_t)c * P_TOT + pbase;
#pragma unroll
    for (int u = 0; u < 32; ++u) {
      bf16_t val = (u < VV) ? src[u] : (bf16_t)0.f;
      xsA[c * 32 + u] = val;
      xsB[bswz(c, u, 2)] = val;
    }
  }
  for (int i = tid; i < SS * 32 * 32; i += 256) {  // att (already zero-padded)
    int s = i >> 10, rem = i & 1023;
    int u = rem >> 5, v = rem & 31;
    attsB[s * 1024 + bswz(u, v, 2)] =
        (bf16_t)attp[((size_t)(n * SS + s) * 32 + u) * 32 + v];
  }
  __syncthreads();

  // ---- stage 1: y2[s*256+c][v] = sum_u x[c][u] * att_s[u][v]  (96 tiles / 8 waves)
#pragma unroll 1
  for (int tt = wave; tt < 96; tt += 8) {
    int s = tt / 32, rem = tt % 32, mt = rem >> 1, nt = rem & 1;
    int cbase = mt * 16;
    v16bf a = a_frag_lds32(xsA, cbase, 0);
    v16bf b = b_frag_swz(attsB + s * 1024, 0, nt, 2);
    FragF32 acc; zero8(acc);
    acc.v = wmma_bf16(a, b, acc.v);
    int col = (lane & 15) + nt * 16;
#pragma unroll
    for (int r = 0; r < 8; ++r) {
      int row = s * 256 + cbase + r + ((lane >> 4) << 3);
      y2sB[bswz(row, col, 2)] = (bf16_t)acc.e[r];
    }
  }
  __syncthreads();

  // ---- stage 2: out_w(256x768)@y2 and d1_w(256x256)@x; wave owns 32 output rows
  int Mbase = wave * 32;
  FragF32 accO[2][2], accD[2][2];
#pragma unroll
  for (int mt = 0; mt < 2; ++mt)
#pragma unroll
    for (int nt = 0; nt < 2; ++nt) { zero8(accO[mt][nt]); zero8(accD[mt][nt]); }
#pragma unroll 1
  for (int kk = 0; kk < 24; ++kk) {
    v16bf b0 = b_frag_swz(y2sB, kk, 0, 2);
    v16bf b1 = b_frag_swz(y2sB, kk, 1, 2);
#pragma unroll
    for (int mt = 0; mt < 2; ++mt) {
      __builtin_prefetch(outw + (size_t)(Mbase + mt * 16) * 768 + kk * 32 + 32, 0, 0);
      v16bf a = a_frag_rowmajor(outw, Mbase + mt * 16, 768, kk * 32);
      accO[mt][0].v = wmma_bf16(a, b0, accO[mt][0].v);
      accO[mt][1].v = wmma_bf16(a, b1, accO[mt][1].v);
    }
  }
#pragma unroll 1
  for (int kk = 0; kk < 8; ++kk) {
    v16bf b0 = b_frag_swz(xsB, kk, 0, 2);
    v16bf b1 = b_frag_swz(xsB, kk, 1, 2);
#pragma unroll
    for (int mt = 0; mt < 2; ++mt) {
      v16bf a = a_frag_rowmajor(d1w, Mbase + mt * 16, 256, kk * 32);
      accD[mt][0].v = wmma_bf16(a, b0, accD[mt][0].v);
      accD[mt][1].v = wmma_bf16(a, b1, accD[mt][1].v);
    }
  }
  // u = lrelu(bn_out(out) + bn_d1(d1))
  float uval[2][2][8];
#pragma unroll
  for (int mt = 0; mt < 2; ++mt)
#pragma unroll
    for (int nt = 0; nt < 2; ++nt)
#pragma unroll
      for (int r = 0; r < 8; ++r) {
        int o = Mbase + mt * 16 + r + ((lane >> 4) << 3);
        float z = accO[mt][nt].e[r] * scl[0 * 256 + o] + bia[0 * 256 + o] +
                  accD[mt][nt].e[r] * scl[1 * 256 + o] + bia[1 * 256 + o];
        uval[mt][nt][r] = (z >= 0.f) ? z : 0.1f * z;
      }
  __syncthreads();             // all waves done reading y2sB
  bf16_t* usB = y2sB;          // alias: u activation, swizzled K=256
#pragma unroll
  for (int mt = 0; mt < 2; ++mt)
#pragma unroll
    for (int nt = 0; nt < 2; ++nt)
#pragma unroll
      for (int r = 0; r < 8; ++r) {
        int o = Mbase + mt * 16 + r + ((lane >> 4) << 3);
        int col = nt * 16 + (lane & 15);
        usB[bswz(o, col, 2)] = (bf16_t)uval[mt][nt][r];
      }
  __syncthreads();

  // ---- stage 3: ff_w@u and d2_w@x ; y = lrelu(bn_d2(d2) + lrelu(bn_ff(ff)))
  FragF32 accF[2][2], accG[2][2];
#pragma unroll
  for (int mt = 0; mt < 2; ++mt)
#pragma unroll
    for (int nt = 0; nt < 2; ++nt) { zero8(accF[mt][nt]); zero8(accG[mt][nt]); }
#pragma unroll 1
  for (int kk = 0; kk < 8; ++kk) {
    v16bf bu0 = b_frag_swz(usB, kk, 0, 2);
    v16bf bu1 = b_frag_swz(usB, kk, 1, 2);
    v16bf bx0 = b_frag_swz(xsB, kk, 0, 2);
    v16bf bx1 = b_frag_swz(xsB, kk, 1, 2);
#pragma unroll
    for (int mt = 0; mt < 2; ++mt) {
      v16bf af = a_frag_rowmajor(ffw, Mbase + mt * 16, 256, kk * 32);
      v16bf ad = a_frag_rowmajor(d2w, Mbase + mt * 16, 256, kk * 32);
      accF[mt][0].v = wmma_bf16(af, bu0, accF[mt][0].v);
      accF[mt][1].v = wmma_bf16(af, bu1, accF[mt][1].v);
      accG[mt][0].v = wmma_bf16(ad, bx0, accG[mt][0].v);
      accG[mt][1].v = wmma_bf16(ad, bx1, accG[mt][1].v);
    }
  }
#pragma unroll
  for (int mt = 0; mt < 2; ++mt)
#pragma unroll
    for (int nt = 0; nt < 2; ++nt)
#pragma unroll
      for (int r = 0; r < 8; ++r) {
        int o = Mbase + mt * 16 + r + ((lane >> 4) << 3);
        int v = nt * 16 + (lane & 15);
        if (v < VV) {
          float f = accF[mt][nt].e[r] * scl[3 * 256 + o] + bia[3 * 256 + o];
          f = (f >= 0.f) ? f : 0.1f * f;
          float z = accG[mt][nt].e[r] * scl[2 * 256 + o] + bia[2 * 256 + o] + f;
          z = (z >= 0.f) ? z : 0.1f * z;
          out[((size_t)(n * CC + o) * TT + t) * VV + v] = z;
        }
      }
}

// ---------------- host launcher ----------------
extern "C" void kernel_launch(void* const* d_in, const int* in_sizes, int n_in,
                              void* d_out, int out_size, void* d_ws, size_t ws_size,
                              hipStream_t stream) {
  (void)in_sizes; (void)n_in; (void)out_size; (void)ws_size;
  const float* x      = (const float*)d_in[0];
  const float* in_w   = (const float*)d_in[1];
  const float* in_b   = (const float*)d_in[2];
  const float* alphas = (const float*)d_in[3];
  const float* att0   = (const float*)d_in[4];
  const float* out_w  = (const float*)d_in[5];
  const float* out_b  = (const float*)d_in[6];
  const float* d1_w   = (const float*)d_in[7];
  const float* d1_b   = (const float*)d_in[8];
  const float* d2_w   = (const float*)d_in[9];
  const float* d2_b   = (const float*)d_in[10];
  const float* ff_w   = (const float*)d_in[11];
  const float* ff_b   = (const float*)d_in[12];
  const float* out_g  = (const float*)d_in[13];
  const float* out_be = (const float*)d_in[14];
  const float* out_m  = (const float*)d_in[15];
  const float* out_v  = (const float*)d_in[16];
  const float* d1_g   = (const float*)d_in[17];
  const float* d1_be  = (const float*)d_in[18];
  const float* d1_m   = (const float*)d_in[19];
  const float* d1_v   = (const float*)d_in[20];
  const float* d2_g   = (const float*)d_in[21];
  const float* d2_be  = (const float*)d_in[22];
  const float* d2_m   = (const float*)d_in[23];
  const float* d2_v   = (const float*)d_in[24];
  const float* ff_g   = (const float*)d_in[25];
  const float* ff_be  = (const float*)d_in[26];
  const float* ff_m   = (const float*)d_in[27];
  const float* ff_v   = (const float*)d_in[28];

  char* w = (char*)d_ws;
  auto take = [&](size_t bytes) {
    char* p = w;
    w += (bytes + 255) & ~(size_t)255;
    return (void*)p;
  };
  bf16_t* Xbf  = (bf16_t*)take((size_t)CC * P_TOT * 2);
  bf16_t* Abf  = (bf16_t*)take((size_t)CC * P_TOT * 2);
  bf16_t* QK   = (bf16_t*)take((size_t)(2 * SS * ICC) * P_TOT * 2);
  float*  attp = (float*) take((size_t)NN * SS * 32 * 32 * 4);
  bf16_t* inw  = (bf16_t*)take((size_t)384 * 256 * 2);
  bf16_t* outw = (bf16_t*)take((size_t)256 * 768 * 2);
  bf16_t* d1w  = (bf16_t*)take((size_t)256 * 256 * 2);
  bf16_t* d2w  = (bf16_t*)take((size_t)256 * 256 * 2);
  bf16_t* ffw  = (bf16_t*)take((size_t)256 * 256 * 2);
  float*  scl  = (float*) take(1024 * 4);
  float*  bia  = (float*) take(1024 * 4);

  prep_acts<<<4096, 256, 0, stream>>>(x, Xbf, Abf);
  prep_weights<<<1920, 256, 0, stream>>>(
      in_w, out_w, d1_w, d2_w, ff_w,
      out_b, d1_b, d2_b, ff_b,
      out_g, out_be, out_m, out_v,
      d1_g, d1_be, d1_m, d1_v,
      d2_g, d2_be, d2_m, d2_v,
      ff_g, ff_be, ff_m, ff_v,
      inw, outw, d1w, d2w, ffw, scl, bia);
  qk_gemm<<<3 * (P_TOT / 64), 256, 0, stream>>>(Abf, inw, in_b, QK);
  att_gemm<<<(NN * SS) / 8, 256, 0, stream>>>(QK, alphas, att0, attp);
  fused_out<<<NN * TT, 256, 0, stream>>>(Xbf, attp, outw, d1w, d2w, ffw, scl, bia,
                                         (float*)d_out);
}
